// SelfAttention_79113297592342
// MI455X (gfx1250) — matrix operand order repaired and checked by reading, not compiled, not run on hardware
//
#include <hip/hip_runtime.h>

// ---------------------------------------------------------------------------
// Problem constants (from reference): B=4, L=1024, HID=768, H=12, D=64
// ---------------------------------------------------------------------------
#define Bc   4
#define Lc   1024
#define Hc   12
#define Dc   64
#define HIDc 768
#define BHc  (Bc*Hc)          // 48
#define Mrows (Bc*Lc)         // 4096

#define DEVINL __device__ __forceinline__

typedef __attribute__((ext_vector_type(16))) __bf16 v16bf;
typedef __attribute__((ext_vector_type(8)))  __bf16 v8bf;
typedef __attribute__((ext_vector_type(8)))  float  v8f;
typedef __attribute__((ext_vector_type(4)))  unsigned int v4u;
typedef __attribute__((ext_vector_type(8)))  int v8i;
typedef __attribute__((ext_vector_type(4)))  int v4i;
typedef unsigned int u32;
typedef unsigned long long u64;

union V16U { v16bf v; v8bf h[2]; };

DEVINL v8f wmma_bf16(v16bf a, v16bf b, v8f c) {
  // D = A(16x32 bf16) x B(32x16 bf16) + C(16x16 f32)
  return __builtin_amdgcn_wmma_f32_16x16x32_bf16(false, a, false, b, (short)0, c,
                                                 false, false);
}

// A-fragment (16x32 bf16). Lane (h = lane>>4, lm = lane&15) holds row `rowPtr`.
// elements j=0..7  -> k = k0 + h*8 + j          (VGPRs 0-3 of the ISA layout)
// elements j=8..15 -> k = k0 + 16 + h*8 + (j-8) (VGPRs 4-7)
DEVINL v16bf load_a(const __bf16* rowPtr, int k0, int h) {
  V16U u;
  u.h[0] = *(const v8bf*)(rowPtr + k0 + h * 8);
  u.h[1] = *(const v8bf*)(rowPtr + k0 + 16 + h * 8);
  return u.v;
}

// B-fragment (32x16 bf16). Lane column = lm; element j<8 -> k = k0 + h*8 + j, j>=8 -> k = k0 + 16 + h*8 + (j-8).
DEVINL v16bf load_b(const __bf16* colPtr, int k0, int h) {
  V16U u;
  u.h[0] = *(const v8bf*)(colPtr + k0 + h * 8);
  u.h[1] = *(const v8bf*)(colPtr + k0 + 16 + h * 8);
  return u.v;
}

// Same as load_b but converting from an fp32 source (global or LDS).
DEVINL v16bf load_b_f32(const float* colPtr, int k0, int h) {
  const float4* p = (const float4*)(colPtr + k0 + h * 8);
  v16bf r;
#pragma unroll
  for (int q = 0; q < 4; ++q) {
    float4 f = p[(q & 1) + (q >> 1) * 4];
    r[4 * q + 0] = (__bf16)f.x;
    r[4 * q + 1] = (__bf16)f.y;
    r[4 * q + 2] = (__bf16)f.z;
    r[4 * q + 3] = (__bf16)f.w;
  }
  return r;
}

// ---------------------------------------------------------------------------
// Tensor Data Mover: 2D tile (tile_d1 lines of tile_d0 fp32 elements,
// line stride stride_elems) from global -> LDS.  D# per ISA ch.8.
// This toolchain exposes the 6-arg builtin (extra int32x8 group before cpol).
// ---------------------------------------------------------------------------
DEVINL void tdm_load_2d_f32(u32 lds_addr, const void* gptr, u32 tile_d0,
                            u32 tile_d1, u64 stride_elems, u32 tensor_d0,
                            u32 tensor_d1) {
  const u64 ga = (u64)(uintptr_t)gptr;
  v4u g0;
  g0[0] = 1u;                                    // count=1 (valid), user mode
  g0[1] = lds_addr;                              // LDS byte address
  g0[2] = (u32)ga;                               // global_addr[31:0]
  g0[3] = ((u32)(ga >> 32) & 0x1FFFFFFu) | (2u << 30);  // [56:32] | type=2
  v8i g1;
  g1[0] = (int)(2u << 16);                       // data_size=2 -> 4 bytes
  g1[1] = (int)((tensor_d0 & 0xFFFFu) << 16);    // tensor_dim0[15:0] @bit48
  g1[2] = (int)((tensor_d0 >> 16) | ((tensor_d1 & 0xFFFFu) << 16));
  g1[3] = (int)((tensor_d1 >> 16) | (tile_d0 << 16));   // tile_dim0 @bit112
  g1[4] = (int)(tile_d1 & 0xFFFFu);              // tile_dim1 (tile_dim2=0)
  g1[5] = (int)(u32)(stride_elems & 0xFFFFFFFFu);       // dim0_stride lo
  g1[6] = (int)(u32)((stride_elems >> 32) & 0xFFFFu);   // dim0_stride hi
  g1[7] = 0;                                     // tensor_dim1_stride (unused)
  v4i z4 = {0, 0, 0, 0};
  v8i z8 = {0, 0, 0, 0, 0, 0, 0, 0};
  __builtin_amdgcn_tensor_load_to_lds(g0, g1, z4, z4, z8, 0);
}

// ---------------------------------------------------------------------------
// Kernel 1: fp32 -> bf16 convert (hidden_states and the three weights)
// ---------------------------------------------------------------------------
__global__ void k_convert(const float* __restrict__ X,
                          const float* __restrict__ Wq,
                          const float* __restrict__ Wk,
                          const float* __restrict__ Wv,
                          unsigned short* __restrict__ Xbf_u,
                          unsigned short* __restrict__ Wbf_u) {
  __bf16* Xbf = (__bf16*)Xbf_u;
  __bf16* Wbf = (__bf16*)Wbf_u;
  const int XE = Mrows * HIDc;
  const int WE = HIDc * HIDc;
  const int total = XE + 3 * WE;
  for (int i = blockIdx.x * blockDim.x + threadIdx.x; i < total;
       i += gridDim.x * blockDim.x) {
    if (i < XE) {
      Xbf[i] = (__bf16)X[i];
    } else {
      int j = i - XE;
      int w = j / WE, o = j - w * WE;
      const float* Ws = (w == 0) ? Wq : ((w == 1) ? Wk : Wv);
      Wbf[j] = (__bf16)Ws[o];
    }
  }
}

// ---------------------------------------------------------------------------
// Kernel 2: QKV projection  C[m,n] = sum_k X[m,k] * W[n,k] + bias[n]
// grid (Mrows/16, HID/16, 3), block 32 (one wave per 16x16 tile).
// Q,K stored [bh][L][D] bf16 row-major; V stored transposed [bh][D][L] bf16.
// ---------------------------------------------------------------------------
__global__ void k_qkv_proj(const unsigned short* __restrict__ Xbf_u,
                           const unsigned short* __restrict__ Wbf_u,
                           const float* __restrict__ bq,
                           const float* __restrict__ bk,
                           const float* __restrict__ bv,
                           unsigned short* __restrict__ Qbf_u,
                           unsigned short* __restrict__ Kbf_u,
                           unsigned short* __restrict__ Vt_u) {
  const __bf16* Xbf = (const __bf16*)Xbf_u;
  const __bf16* Wbf = (const __bf16*)Wbf_u;
  const int lane = threadIdx.x & 31;
  const int h = lane >> 4, lm = lane & 15;
  const int m0 = blockIdx.x * 16;
  const int n0 = blockIdx.y * 16;
  const int w = blockIdx.z;

  const __bf16* W = Wbf + (size_t)w * HIDc * HIDc;
  const float* bias = (w == 0) ? bq : ((w == 1) ? bk : bv);

  const __bf16* arow = Xbf + (size_t)(m0 + lm) * HIDc;
  const __bf16* brow = W + (size_t)(n0 + lm) * HIDc;

  v8f c = {};
#pragma unroll 4
  for (int k0 = 0; k0 < HIDc; k0 += 32) {
    v16bf a = load_a(arow, k0, h);
    v16bf b = load_b(brow, k0, h);
    c = wmma_bf16(a, b, c);
  }

  const float bn = bias[n0 + lm];
  const int bidx = m0 >> 10;          // batch (tiles never cross L boundary)
  const int head = n0 >> 6;           // head (16-wide tile never crosses 64)
  const int d = (n0 & 63) + lm;

  __bf16* Qbf = (__bf16*)Qbf_u;
  __bf16* Kbf = (__bf16*)Kbf_u;
  __bf16* Vt = (__bf16*)Vt_u;

#pragma unroll
  for (int j = 0; j < 8; ++j) {
    const int m = m0 + j + h * 8;
    const int l = m & (Lc - 1);
    const __bf16 vb = (__bf16)(c[j] + bn);
    const size_t bh = (size_t)(bidx * Hc + head);
    if (w == 0)      Qbf[(bh * Lc + l) * Dc + d] = vb;
    else if (w == 1) Kbf[(bh * Lc + l) * Dc + d] = vb;
    else             Vt[(bh * Dc + d) * Lc + l] = vb;
  }
}

// ---------------------------------------------------------------------------
// Kernel 3 (fused biases): one block per 16x16 (l,r) tile of S -> S read ONCE.
//   BiasQ[bh][l][r] = sum_d Q[bh,l,d] * S[l,r,d]  (GEMM for fixed l)
//   BiasK[bh][r][l] = sum_d K[bh,r,d] * S[l,r,d]  (GEMM for fixed r)
// The 64KB fp32 S tile is DMA'd to LDS by the Tensor Data Mover (one wave
// issues it behind a scalar branch; TDM ignores EXEC so a lane-mask guard
// would issue 8 duplicate DMAs). 8 waves then split 32 GEMM jobs.
// ---------------------------------------------------------------------------
__global__ void k_bias_fused(const unsigned short* __restrict__ Qbf_u,
                             const unsigned short* __restrict__ Kbf_u,
                             const float* __restrict__ S,
                             unsigned short* __restrict__ BiasQ_u,
                             unsigned short* __restrict__ BiasK_u) {
  __shared__ __align__(16) float Slds[16 * 16 * Dc];  // [l_local][r_local][d]

  const __bf16* Qbf = (const __bf16*)Qbf_u;
  const __bf16* Kbf = (const __bf16*)Kbf_u;
  __bf16* BiasQ = (__bf16*)BiasQ_u;
  __bf16* BiasK = (__bf16*)BiasK_u;

  const int l0 = blockIdx.x * 16;
  const int r0 = blockIdx.y * 16;
  const int lane = threadIdx.x & 31;
  const int h = lane >> 4, lm = lane & 15;
  const int waveid = __builtin_amdgcn_readfirstlane((int)(threadIdx.x >> 5));

  // --- TDM: stage S[l0:l0+16][r0:r0+16][0:64] (16 lines x 4KB) into LDS ---
  if (waveid == 0) {
    const float* gsrc = S + ((size_t)l0 * Lc + r0) * Dc;
    tdm_load_2d_f32((u32)(uintptr_t)&Slds[0], gsrc,
                    /*tile_d0=*/16 * Dc, /*tile_d1=*/16,
                    /*stride=*/(u64)Lc * Dc,
                    /*tensor_d0=*/(u32)Lc * Dc, /*tensor_d1=*/(u32)Lc);
    __builtin_amdgcn_s_wait_tensorcnt(0);
  }
  __syncthreads();

  // --- 32 jobs: 16 bias_q (per l) + 16 bias_k (per r), 4 per wave ---
  for (int job = waveid; job < 32; job += 8) {
    if (job < 16) {
      const int ll = job;                 // bias_q at l = l0 + ll
      const int l = l0 + ll;
      const float* sb = &Slds[(ll * 16 + lm) * Dc];  // lane's column r = lm
      v16bf b0 = load_b_f32(sb, 0, h);
      v16bf b1 = load_b_f32(sb, 32, h);
#pragma unroll
      for (int t = 0; t < 3; ++t) {
        const __bf16* qrow = Qbf + ((size_t)(t * 16 + lm) * Lc + l) * Dc;
        v8f c = {};
        c = wmma_bf16(load_a(qrow, 0, h), b0, c);
        c = wmma_bf16(load_a(qrow, 32, h), b1, c);
#pragma unroll
        for (int j = 0; j < 8; ++j) {
          const int bh = t * 16 + j + h * 8;
          BiasQ[((size_t)bh * Lc + l) * Lc + r0 + lm] = (__bf16)c[j];
        }
      }
    } else {
      const int rl = job - 16;            // bias_k at r = r0 + rl
      const int r = r0 + rl;
      const float* sb = &Slds[(lm * 16 + rl) * Dc];  // lane's column l = lm
      v16bf b0 = load_b_f32(sb, 0, h);
      v16bf b1 = load_b_f32(sb, 32, h);
#pragma unroll
      for (int t = 0; t < 3; ++t) {
        const __bf16* krow = Kbf + ((size_t)(t * 16 + lm) * Lc + r) * Dc;
        v8f c = {};
        c = wmma_bf16(load_a(krow, 0, h), b0, c);
        c = wmma_bf16(load_a(krow, 32, h), b1, c);
#pragma unroll
        for (int j = 0; j < 8; ++j) {
          const int bh = t * 16 + j + h * 8;
          BiasK[((size_t)bh * Lc + r) * Lc + l0 + lm] = (__bf16)c[j];
        }
      }
    }
  }
}

// ---------------------------------------------------------------------------
// Kernel 4: flash attention with structure biases.
// grid (L/16, BH), one wave per block. 32 steps of 32 keys.
// ---------------------------------------------------------------------------
__global__ void k_attn(const unsigned short* __restrict__ Qbf_u,
                       const unsigned short* __restrict__ Kbf_u,
                       const unsigned short* __restrict__ Vt_u,
                       const unsigned short* __restrict__ BiasQ_u,
                       const unsigned short* __restrict__ BiasK_u,
                       const float* __restrict__ mask,
                       float* __restrict__ out) {
  const __bf16* Qbf = (const __bf16*)Qbf_u;
  const __bf16* Kbf = (const __bf16*)Kbf_u;
  const __bf16* Vt = (const __bf16*)Vt_u;
  const __bf16* BiasQ = (const __bf16*)BiasQ_u;
  const __bf16* BiasK = (const __bf16*)BiasK_u;

  __shared__ __bf16 pT[16 * 32];  // probs tile for C->A layout transpose

  const int lane = threadIdx.x & 31;
  const int h = lane >> 4, lm = lane & 15;
  const int l0 = blockIdx.x * 16;
  const int bh = blockIdx.y;
  const int b = bh / Hc, hh = bh - b * Hc;

  // Q fragments for this 16-query tile (K = d = 64)
  const __bf16* qrow = Qbf + ((size_t)bh * Lc + l0 + lm) * Dc;
  const v16bf aq0 = load_a(qrow, 0, h);
  const v16bf aq1 = load_a(qrow, 32, h);

  v8f O[4] = {{}, {}, {}, {}};  // 16 x 64 f32 accumulator (4 d-tiles)
  float mrow[8], lsum[8];
#pragma unroll
  for (int j = 0; j < 8; ++j) { mrow[j] = -1e30f; lsum[j] = 0.f; }

  const float* maskb = mask + (size_t)b * Lc;
  const size_t bhLL = (size_t)bh * Lc;

  for (int r0 = 0; r0 < Lc; r0 += 32) {
    // ---- scores: QK^T for two 16-wide r sub-tiles ----
    v8f c0 = {}, c1 = {};
    {
      const __bf16* kr0 = Kbf + (bhLL + r0 + lm) * Dc;
      const __bf16* kr1 = kr0 + 16 * Dc;
      c0 = wmma_bf16(aq0, load_b(kr0, 0, h), c0);
      c0 = wmma_bf16(aq1, load_b(kr0, 32, h), c0);
      c1 = wmma_bf16(aq0, load_b(kr1, 0, h), c1);
      c1 = wmma_bf16(aq1, load_b(kr1, 32, h), c1);
      if (r0 + 32 < Lc) {  // prefetch next K tile (global_prefetch_b8)
        __builtin_prefetch(kr0 + 32 * Dc, 0, 1);
        __builtin_prefetch(kr1 + 32 * Dc, 0, 1);
      }
    }
    const float mk0 = maskb[r0 + lm];
    const float mk1 = maskb[r0 + 16 + lm];

    float s0[8], s1[8];
#pragma unroll
    for (int j = 0; j < 8; ++j) {
      const int l = l0 + j + h * 8;
      const size_t bq = (bhLL + l) * Lc;                 // BiasQ[bh][l][.]
      const float t0 = c0[j] + (float)BiasQ[bq + r0 + lm] +
                       (float)BiasK[(bhLL + r0 + lm) * Lc + l];
      const float t1 = c1[j] + (float)BiasQ[bq + r0 + 16 + lm] +
                       (float)BiasK[(bhLL + r0 + 16 + lm) * Lc + l];
      s0[j] = t0 * 0.125f + mk0;   // 1/sqrt(64)
      s1[j] = t1 * 0.125f + mk1;
    }

    // ---- online softmax update (row reductions over 16 lanes/half) ----
#pragma unroll
    for (int j = 0; j < 8; ++j) {
      float mx = fmaxf(s0[j], s1[j]);
#pragma unroll
      for (int off = 1; off < 16; off <<= 1)
        mx = fmaxf(mx, __shfl_xor(mx, off, 16));
      const float mnew = fmaxf(mrow[j], mx);
      const float alpha = __expf(mrow[j] - mnew);
      const float p0 = __expf(s0[j] - mnew);
      const float p1 = __expf(s1[j] - mnew);
      float rs = p0 + p1;
#pragma unroll
      for (int off = 1; off < 16; off <<= 1)
        rs += __shfl_xor(rs, off, 16);
      lsum[j] = lsum[j] * alpha + rs;
      mrow[j] = mnew;
#pragma unroll
      for (int t = 0; t < 4; ++t) O[t][j] *= alpha;
      pT[(j + h * 8) * 32 + lm] = (__bf16)p0;
      pT[(j + h * 8) * 32 + 16 + lm] = (__bf16)p1;
    }
    __syncthreads();

    // ---- P (16x32) as A-fragment via LDS transpose ----
    v16bf pa;
    {
      V16U u;
      u.h[0] = *(const v8bf*)(&pT[lm * 32 + h * 8]);
      u.h[1] = *(const v8bf*)(&pT[lm * 32 + 16 + h * 8]);
      pa = u.v;
    }
    __syncthreads();

    // ---- O += P @ V  (V stored transposed [bh][d][L]) ----
#pragma unroll
    for (int t = 0; t < 4; ++t) {
      const __bf16* vrow = Vt + ((size_t)bh * Dc + t * 16 + lm) * Lc + r0;
      O[t] = wmma_bf16(pa, load_b(vrow, 0, h), O[t]);
    }
  }

  // ---- normalize and write ctx -> [B, L, H, D] ----
#pragma unroll
  for (int j = 0; j < 8; ++j) {
    const float inv = 1.f / lsum[j];
    const int l = l0 + j + h * 8;
    float* orow = out + ((size_t)(b * Lc + l) * Hc + hh) * Dc;
#pragma unroll
    for (int t = 0; t < 4; ++t) orow[t * 16 + lm] = O[t][j] * inv;
  }
}

// ---------------------------------------------------------------------------
// Host launcher
// ---------------------------------------------------------------------------
extern "C" void kernel_launch(void* const* d_in, const int* in_sizes, int n_in,
                              void* d_out, int out_size, void* d_ws,
                              size_t ws_size, hipStream_t stream) {
  const float* hidden = (const float*)d_in[0];
  const float* mask = (const float*)d_in[1];
  const float* S = (const float*)d_in[2];
  const float* Wq = (const float*)d_in[3];
  const float* bq = (const float*)d_in[4];
  const float* Wk = (const float*)d_in[5];
  const float* bk = (const float*)d_in[6];
  const float* Wv = (const float*)d_in[7];
  const float* bv = (const float*)d_in[8];
  float* out = (float*)d_out;

  const size_t XE = (size_t)Mrows * HIDc;     // 3.1M
  const size_t WE = (size_t)HIDc * HIDc;      // 0.59M
  const size_t QE = (size_t)BHc * Lc * Dc;    // 3.1M
  const size_t BE = (size_t)BHc * Lc * Lc;    // 50.3M

  char* ws = (char*)d_ws;
  size_t off = 0;
  auto alloc = [&](size_t bytes) {
    void* p = ws + off;
    off += (bytes + 255) & ~(size_t)255;
    return p;
  };
  unsigned short* Xbf = (unsigned short*)alloc(XE * 2);
  unsigned short* Wbf = (unsigned short*)alloc(3 * WE * 2);
  unsigned short* Qbf = (unsigned short*)alloc(QE * 2);
  unsigned short* Kbf = (unsigned short*)alloc(QE * 2);
  unsigned short* Vt = (unsigned short*)alloc(QE * 2);
  unsigned short* BiasQ = (unsigned short*)alloc(BE * 2);
  unsigned short* BiasK = (unsigned short*)alloc(BE * 2);
  (void)in_sizes; (void)n_in; (void)out_size; (void)ws_size;

  // 1) fp32 -> bf16 conversions
  k_convert<<<2048, 256, 0, stream>>>(hidden, Wq, Wk, Wv, Xbf, Wbf);

  // 2) QKV projection (bf16 WMMA GEMM)
  k_qkv_proj<<<dim3(Mrows / 16, HIDc / 16, 3), 32, 0, stream>>>(
      Xbf, Wbf, bq, bk, bv, Qbf, Kbf, Vt);

  // 3) fused structure biases: S streamed exactly once via TDM
  k_bias_fused<<<dim3(Lc / 16, Lc / 16), 256, 0, stream>>>(Qbf, Kbf, S, BiasQ,
                                                           BiasK);

  // 4) flash attention with biases
  k_attn<<<dim3(Lc / 16, BHc), 32, 0, stream>>>(Qbf, Kbf, Vt, BiasQ, BiasK,
                                                mask, out);
}
